// Transformer_5961414607463
// MI455X (gfx1250) — matrix operand order repaired and checked by reading, not compiled, not run on hardware
//
#include <hip/hip_runtime.h>
#include <hip/hip_bf16.h>

// ---------------------------------------------------------------------------
// Restormer-style block for gfx1250 (MI455X), wave32 + WMMA f16.
// GEMMs are templated on (LD, KSTEPS, F32OUT): fully-unrolled K loops where
// every operand load is a global_load_b128 with a compile-time immediate
// offset from one per-lane base pointer -> near-zero address VALU between
// wmmas.  Activations are N-major (K contiguous), K zero-padded to x32.
// ---------------------------------------------------------------------------

typedef __attribute__((ext_vector_type(16))) _Float16 v16h;
typedef __attribute__((ext_vector_type(8)))  float    v8f;

#define BN    4
#define CDIM  192
#define HWSZ  16384          // 128*128
#define HEADS 4
#define CH    48             // channels per head
#define HIDN  510
#define HID2  1020
#define HID2P 1024
#define HIDKP 512
#define VLD   256            // padded row length of transposed V buffers

union V16H { v16h v; _Float16 h[16]; uint4 q[2]; };

#define WMMA_F16(A, B, C) \
  __builtin_amdgcn_wmma_f32_16x16x32_f16(false, (A), false, (B), (short)0, (C), false, false)

// Load 16 halves as two b128s: slots 0-7 at p, slots 8-15 at p+off1.
// A operand (MxK, row-major): per-lane base = &W[row0+(lane&15)][(lane&16)>>1],
//   off1 = 16  (K kb..kb+7 then kb+16..kb+23).
// B operand (KxN from XT[n][k]): per-lane base = &XT[col0+(lane&15)][(lane&16)>>1],
//   off1 = 16  (K kb..kb+7 then kb+16..kb+23).
__device__ __forceinline__ v16h load16(const _Float16* __restrict__ p, int off1) {
  V16H r;
  r.q[0] = *(const uint4*)(p);
  r.q[1] = *(const uint4*)(p + off1);
  return r.v;
}

// ---- weight convert f32 -> f16 with zero padding ---------------------------
__global__ void wconv_k(const float* __restrict__ src, int srows, int scols,
                        _Float16* __restrict__ dst, int drows, int dcols) {
  int gid = blockIdx.x * 256 + threadIdx.x;
  if (gid >= drows * dcols) return;
  int r = gid / dcols, c = gid % dcols;
  float v = (r < srows && c < scols) ? src[r * scols + c] : 0.0f;
  dst[gid] = (_Float16)v;
}

// ---- channel LayerNorm: f32 channel-major in -> f16 TRANSPOSED out ---------
__global__ void ln_k(const float* __restrict__ in, const float* __restrict__ w,
                     const float* __restrict__ bia, _Float16* __restrict__ outT,
                     int ldT) {
  __shared__ float tile[CDIM * 33];
  int p0 = blockIdx.x * 32;                 // 32 pixels per block
  int b  = p0 >> 14;
  int pb = p0 & (HWSZ - 1);
  const float* ip = in + (size_t)b * CDIM * HWSZ + pb;
  int t = threadIdx.x;
#pragma unroll
  for (int i = 0; i < 24; ++i) {
    int idx = t + 256 * i;                  // 0..6143
    int c = idx >> 5, p = idx & 31;
    tile[c * 33 + p] = ip[(size_t)c * HWSZ + p];
  }
  __syncthreads();
  int lane = t & 31, grp = t >> 5;
  for (int pp = grp; pp < 32; pp += 8) {
    float v[6], s = 0.f, ss = 0.f;
#pragma unroll
    for (int i = 0; i < 6; ++i) {
      int c = lane + 32 * i;
      v[i] = tile[c * 33 + pp];
      s += v[i]; ss += v[i] * v[i];
    }
#pragma unroll
    for (int off = 16; off > 0; off >>= 1) {
      s  += __shfl_xor(s,  off, 32);
      ss += __shfl_xor(ss, off, 32);
    }
    float mu   = s * (1.0f / CDIM);
    float var  = ss * (1.0f / CDIM) - mu * mu;
    float rstd = rsqrtf(var + 1e-5f);
    _Float16* op = outT + ((size_t)b * HWSZ + pb + pp) * ldT;
#pragma unroll
    for (int i = 0; i < 6; ++i) {
      int c = lane + 32 * i;
      op[c] = (_Float16)((v[i] - mu) * rstd * w[c] + bia[c]);
    }
  }
}

// ---- templated WMMA GEMM: Out[b] = W(O x LD) * XT[b](HW x LD)^T (+res) -----
// 8 waves/block, each wave one 64x16 tile.  Fully unrolled, software
// pipelined K loop; all loads use compile-time immediate offsets.
template <int LD, int KSTEPS, bool F32OUT>
__global__ void gemm_k(const _Float16* __restrict__ W,
                       const _Float16* __restrict__ XT,
                       _Float16* __restrict__ outH, float* __restrict__ outF,
                       const float* __restrict__ res, int orows) {
  int lane = threadIdx.x & 31;
  int wid  = threadIdx.x >> 5;
  int b    = blockIdx.z;
  int col0 = (blockIdx.x * 8 + wid) * 16;
  int row0 = blockIdx.y * 64;
  const _Float16* pb = XT + (size_t)b * HWSZ * LD
                          + (size_t)(col0 + (lane & 15)) * LD + ((lane & 16) >> 1);
  const _Float16* pa = W + (size_t)(row0 + (lane & 15)) * LD + ((lane & 16) >> 1);

  v8f acc[4] = {};
  v16h bc = load16(pb, 16);
  v16h a0 = load16(pa,           16);
  v16h a1 = load16(pa + 16 * LD, 16);
  v16h a2 = load16(pa + 32 * LD, 16);
  v16h a3 = load16(pa + 48 * LD, 16);
#pragma unroll
  for (int ks = 1; ks < KSTEPS; ++ks) {
    const int k0 = ks * 32;
    v16h bn = load16(pb + k0, 16);
    v16h n0 = load16(pa + k0,           16);
    v16h n1 = load16(pa + 16 * LD + k0, 16);
    v16h n2 = load16(pa + 32 * LD + k0, 16);
    v16h n3 = load16(pa + 48 * LD + k0, 16);
    // keep the B stream warm one k-step ahead (global_prefetch_b8; speculative)
    __builtin_prefetch(pb + k0 + 32, 0, 0);
    acc[0] = WMMA_F16(a0, bc, acc[0]);
    acc[1] = WMMA_F16(a1, bc, acc[1]);
    acc[2] = WMMA_F16(a2, bc, acc[2]);
    acc[3] = WMMA_F16(a3, bc, acc[3]);
    bc = bn; a0 = n0; a1 = n1; a2 = n2; a3 = n3;
  }
  acc[0] = WMMA_F16(a0, bc, acc[0]);
  acc[1] = WMMA_F16(a1, bc, acc[1]);
  acc[2] = WMMA_F16(a2, bc, acc[2]);
  acc[3] = WMMA_F16(a3, bc, acc[3]);

  int nn = col0 + (lane & 15);
  int mb = (lane & 16) >> 1;
#pragma unroll
  for (int t = 0; t < 4; ++t)
#pragma unroll
    for (int j = 0; j < 8; ++j) {
      int row = row0 + t * 16 + mb + j;
      size_t idx = ((size_t)b * orows + row) * HWSZ + nn;
      if constexpr (F32OUT) outF[idx] = acc[t][j] + res[idx];
      else                  outH[idx] = (_Float16)acc[t][j];
    }
}

// ---- grouped 3x3 conv for q (2 ch/group), channel-major in/out -------------
__global__ void dwconv_q_k(const _Float16* __restrict__ in, const float* __restrict__ w,
                           _Float16* __restrict__ out) {
  size_t gid = (size_t)blockIdx.x * 256 + threadIdx.x;
  int p = (int)(gid % HWSZ);
  size_t r = gid / HWSZ;
  int c = (int)(r % 384);
  int b = (int)(r / 384);
  int y = p >> 7, x = p & 127;
  int cb = (c >> 1) << 1;                   // group base (2 in-ch per group)
  float s = 0.f;
  for (int i = 0; i < 2; ++i) {
    const _Float16* ip = in + ((size_t)b * 384 + cb + i) * HWSZ;
    const float* wp = w + ((size_t)c * 2 + i) * 9;
#pragma unroll
    for (int ky = 0; ky < 3; ++ky) {
      int yy = y + ky - 1;
      if (yy < 0 || yy > 127) continue;
#pragma unroll
      for (int kx = 0; kx < 3; ++kx) {
        int xx = x + kx - 1;
        if (xx < 0 || xx > 127) continue;
        s += wp[ky * 3 + kx] * (float)ip[yy * 128 + xx];
      }
    }
  }
  out[((size_t)b * 384 + c) * HWSZ + p] = (_Float16)s;
}

// ---- depthwise 3x3 for kv: k-half -> channel-major, v-half -> N-major ------
__global__ void dwconv_kv_k(const _Float16* __restrict__ in, const float* __restrict__ w,
                            _Float16* __restrict__ kOut, _Float16* __restrict__ vT) {
  size_t gid = (size_t)blockIdx.x * 256 + threadIdx.x;
  int p = (int)(gid % HWSZ);
  size_t r = gid / HWSZ;
  int c = (int)(r % 384);
  int b = (int)(r / 384);
  int y = p >> 7, x = p & 127;
  const _Float16* ip = in + ((size_t)b * 384 + c) * HWSZ;
  const float* wp = w + (size_t)c * 9;
  float s = 0.f;
#pragma unroll
  for (int ky = 0; ky < 3; ++ky) {
    int yy = y + ky - 1;
    if (yy < 0 || yy > 127) continue;
#pragma unroll
    for (int kx = 0; kx < 3; ++kx) {
      int xx = x + kx - 1;
      if (xx < 0 || xx > 127) continue;
      s += wp[ky * 3 + kx] * (float)ip[yy * 128 + xx];
    }
  }
  if (c < CDIM)
    kOut[((size_t)b * CDIM + c) * HWSZ + p] = (_Float16)s;
  else
    vT[((size_t)b * HWSZ + p) * VLD + (c - CDIM)] = (_Float16)s;
}

// ---- per-row L2 inverse norms for q / k ------------------------------------
__global__ void rownorm_k(const _Float16* __restrict__ qd,
                          const _Float16* __restrict__ k1d,
                          const _Float16* __restrict__ k2d,
                          float* __restrict__ invq, float* __restrict__ invk) {
  int id = blockIdx.x;
  int r = id % CDIM; id /= CDIM;
  int qk = id & 1;   id >>= 1;
  int b = id & 3;
  int a = id >> 2;
  const _Float16* mat = (qk == 0)
      ? qd + ((size_t)b * 384 + a * CDIM + r) * HWSZ
      : (a ? k2d : k1d) + ((size_t)b * CDIM + r) * HWSZ;
  float ss = 0.f;
  for (int i = threadIdx.x; i < HWSZ; i += 256) {
    float v = (float)mat[i];
    ss += v * v;
  }
#pragma unroll
  for (int off = 16; off > 0; off >>= 1) ss += __shfl_xor(ss, off, 32);
  __shared__ float red[8];
  if ((threadIdx.x & 31) == 0) red[threadIdx.x >> 5] = ss;
  __syncthreads();
  if (threadIdx.x == 0) {
    float t = 0.f;
    for (int i = 0; i < 8; ++i) t += red[i];
    float inv = 1.0f / fmaxf(sqrtf(t), 1e-12f);
    (qk ? invk : invq)[(a * 4 + b) * CDIM + r] = inv;
  }
}

// ---- Gram: S = temp * (q/|q|)(k/|k|)^T ; 8-wave K-split + LDS reduce -------
// Pointer-increment K loop, 4x unrolled: 16 b128 loads (immediate offsets)
// then 4 accumulator-chained wmmas per iteration.
__global__ void gram_k(const _Float16* __restrict__ qd,
                       const _Float16* __restrict__ k1d,
                       const _Float16* __restrict__ k2d,
                       const float* __restrict__ invq, const float* __restrict__ invk,
                       const float* __restrict__ temp1, const float* __restrict__ temp2,
                       float* __restrict__ S) {
  __shared__ float part[8 * 256];
  int lane = threadIdx.x & 31;
  int wid  = threadIdx.x >> 5;
  int z = blockIdx.z;
  int a = z >> 4, b = (z >> 2) & 3, h = z & 3;
  const _Float16* qa = qd + ((size_t)b * 384 + a * CDIM + h * CH) * HWSZ;
  const _Float16* ka = (a ? k2d : k1d) + ((size_t)b * CDIM + h * CH) * HWSZ;
  int row0 = blockIdx.x * 16, col0 = blockIdx.y * 16;
  const _Float16* pq = qa + (size_t)(row0 + (lane & 15)) * HWSZ
                          + ((lane & 16) >> 1) + wid * 32;
  const _Float16* pk = ka + (size_t)(col0 + (lane & 15)) * HWSZ
                          + ((lane & 16) >> 1) + wid * 32;
  v8f acc = {};
#pragma unroll 1
  for (int it = 0; it < HWSZ / 1024; ++it) {       // 4 k-steps per iteration
    v16h qa0 = load16(pq,       16);
    v16h ka0 = load16(pk,       16);
    v16h qa1 = load16(pq + 256, 16);
    v16h ka1 = load16(pk + 256, 16);
    v16h qa2 = load16(pq + 512, 16);
    v16h ka2 = load16(pk + 512, 16);
    v16h qa3 = load16(pq + 768, 16);
    v16h ka3 = load16(pk + 768, 16);
    acc = WMMA_F16(qa0, ka0, acc);
    acc = WMMA_F16(qa1, ka1, acc);
    acc = WMMA_F16(qa2, ka2, acc);
    acc = WMMA_F16(qa3, ka3, acc);
    pq += 1024; pk += 1024;
  }
#pragma unroll
  for (int j = 0; j < 8; ++j) part[wid * 256 + lane * 8 + j] = acc[j];
  __syncthreads();
  if (wid == 0) {
    float tp = (a ? temp2 : temp1)[h];
    int nn = col0 + (lane & 15);
    int mb = (lane & 16) >> 1;
    float ik = invk[(a * 4 + b) * CDIM + h * CH + nn];
#pragma unroll
    for (int j = 0; j < 8; ++j) {
      float sum = 0.f;
#pragma unroll
      for (int ww = 0; ww < 8; ++ww) sum += part[ww * 256 + lane * 8 + j];
      int m = row0 + mb + j;
      float iq = invq[(a * 4 + b) * CDIM + h * CH + m];
      S[((size_t)z * CH + m) * CH + nn] = sum * tp * iq * ik;
    }
  }
}

// ---- row softmax 48-wide -> f16 attention matrix, zero-padded to 48x64 -----
__global__ void softmax_k(const float* __restrict__ S, _Float16* __restrict__ A16) {
  int t = blockIdx.x * 256 + threadIdx.x;
  if (t >= 32 * CH) return;
  const float* row = S + (size_t)t * CH;
  _Float16* ar = A16 + (size_t)t * 64;
  float mx = -3.4e38f;
  for (int d = 0; d < CH; ++d) mx = fmaxf(mx, row[d]);
  float sum = 0.f;
  for (int d = 0; d < CH; ++d) sum += expf(row[d] - mx);
  float rs = 1.0f / sum;
  for (int d = 0; d < CH; ++d) ar[d] = (_Float16)(expf(row[d] - mx) * rs);
  for (int d = CH; d < 64; ++d) ar[d] = (_Float16)0.0f;
}

// ---- attention apply: o = A(48x64, zero-pad) * vT^T ; transposed output ----
__global__ void apply_k(const _Float16* __restrict__ A16,
                        const _Float16* __restrict__ v1T,
                        const _Float16* __restrict__ v2T,
                        _Float16* __restrict__ oT) {
  int lane = threadIdx.x & 31;
  int wid  = threadIdx.x >> 5;
  int z = blockIdx.z;
  int a = z >> 4, b = (z >> 2) & 3, h = z & 3;
  int row0 = blockIdx.y * 16;
  int col0 = (blockIdx.x * 8 + wid) * 16;
  const _Float16* pa = A16 + (size_t)z * CH * 64
                           + (size_t)(row0 + (lane & 15)) * 64 + ((lane & 16) >> 1);
  const _Float16* pv = (a ? v2T : v1T) + (size_t)b * HWSZ * VLD + h * CH
                           + (size_t)(col0 + (lane & 15)) * VLD + ((lane & 16) >> 1);
  v16h am0 = load16(pa,      16);
  v16h am1 = load16(pa + 32, 16);
  v16h bm0 = load16(pv,      16);      // pad rows are 0.0
  v16h bm1 = load16(pv + 32, 16);
  v8f acc = {};
  acc = WMMA_F16(am0, bm0, acc);
  acc = WMMA_F16(am1, bm1, acc);
  int nn = col0 + (lane & 15);
  int mb = (lane & 16) >> 1;
  // 8 consecutive channels per lane -> single b128 store into N-major output
  V16H pk;
#pragma unroll
  for (int j = 0; j < 8; ++j) pk.h[j] = (_Float16)acc[j];
  _Float16* op = oT + ((size_t)b * HWSZ + nn) * 384 + a * CDIM + h * CH + row0 + mb;
  *(uint4*)op = pk.q[0];
}

// ---- FFN: fused depthwise 3x3 + exact-GELU gate -> transposed, K-padded ----
__global__ void ffngate_k(const _Float16* __restrict__ hh, const float* __restrict__ w,
                          _Float16* __restrict__ gT) {
  size_t gid = (size_t)blockIdx.x * 256 + threadIdx.x;
  int p = (int)(gid % HWSZ);
  size_t r = gid / HWSZ;
  int c = (int)(r % HIDKP);
  int b = (int)(r / HIDKP);
  _Float16* op = gT + ((size_t)b * HWSZ + p) * HIDKP + c;
  if (c >= HIDN) { *op = (_Float16)0.0f; return; }      // zero K-pad columns
  int y = p >> 7, x = p & 127;
  const _Float16* ip1 = hh + ((size_t)b * HID2P + c) * HWSZ;
  const _Float16* ip2 = hh + ((size_t)b * HID2P + c + HIDN) * HWSZ;
  const float* w1 = w + (size_t)c * 9;
  const float* w2 = w + (size_t)(c + HIDN) * 9;
  float s1 = 0.f, s2 = 0.f;
#pragma unroll
  for (int ky = 0; ky < 3; ++ky) {
    int yy = y + ky - 1;
    if (yy < 0 || yy > 127) continue;
#pragma unroll
    for (int kx = 0; kx < 3; ++kx) {
      int xx = x + kx - 1;
      if (xx < 0 || xx > 127) continue;
      s1 += w1[ky * 3 + kx] * (float)ip1[yy * 128 + xx];
      s2 += w2[ky * 3 + kx] * (float)ip2[yy * 128 + xx];
    }
  }
  float gl = 0.5f * s1 * (1.0f + erff(s1 * 0.70710678118654752f));
  *op = (_Float16)(gl * s2);
}

// ---------------------------------------------------------------------------
extern "C" void kernel_launch(void* const* d_in, const int* in_sizes, int n_in,
                              void* d_out, int out_size, void* d_ws, size_t ws_size,
                              hipStream_t stream) {
  (void)in_sizes; (void)n_in; (void)out_size; (void)ws_size;

  const float* x      = (const float*)d_in[0];
  const float* kv1_in = (const float*)d_in[1];
  const float* kv2_in = (const float*)d_in[2];
  const float* n1w = (const float*)d_in[3],  *n1b = (const float*)d_in[4];
  const float* nk1w = (const float*)d_in[5], *nk1b = (const float*)d_in[6];
  const float* nk2w = (const float*)d_in[7], *nk2b = (const float*)d_in[8];
  const float* n2w = (const float*)d_in[9],  *n2b = (const float*)d_in[10];
  const float* qw    = (const float*)d_in[11];
  const float* kv1w  = (const float*)d_in[12];
  const float* kv2w  = (const float*)d_in[13];
  const float* qdw   = (const float*)d_in[14];
  const float* k1dw  = (const float*)d_in[15];
  const float* k2dw  = (const float*)d_in[16];
  const float* projw = (const float*)d_in[17];
  const float* temp1 = (const float*)d_in[18];
  const float* temp2 = (const float*)d_in[19];
  const float* pinw  = (const float*)d_in[20];
  const float* dww   = (const float*)d_in[21];
  const float* poutw = (const float*)d_in[22];
  float* outp = (float*)d_out;

  // ---- workspace layout (256B aligned) ----
  char* ws = (char*)d_ws;
  size_t off = 0;
  auto alloc = [&](size_t bytes) {
    size_t o = off;
    off += (bytes + 255) & ~(size_t)255;
    return o;
  };
  const size_t SZ_CT16  = (size_t)BN * HWSZ * CDIM * 2;   // N-major, ld=192
  const size_t SZ_C2CM  = (size_t)BN * 384 * HWSZ * 2;    // channel-major 384
  const size_t SZ_VT    = (size_t)BN * HWSZ * VLD * 2;    // N-major V, ld=256

  size_t o_qw16  = alloc((size_t)384 * 192 * 2);
  size_t o_k1w16 = alloc((size_t)384 * 192 * 2);
  size_t o_k2w16 = alloc((size_t)384 * 192 * 2);
  size_t o_pjw16 = alloc((size_t)192 * 384 * 2);
  size_t o_pnw16 = alloc((size_t)HID2P * 192 * 2);
  size_t o_pow16 = alloc((size_t)192 * HIDKP * 2);
  size_t o_invq  = alloc((size_t)2 * 4 * CDIM * 4);
  size_t o_invk  = alloc((size_t)2 * 4 * CDIM * 4);
  size_t o_S     = alloc((size_t)32 * CH * CH * 4);
  size_t o_A16   = alloc((size_t)32 * CH * 64 * 2);
  size_t o_xnT   = alloc(SZ_CT16);                        // reused as y16T
  size_t o_k1nT  = alloc(SZ_CT16);
  size_t o_k2nT  = alloc(SZ_CT16);
  size_t o_qd    = alloc(SZ_C2CM);                        // channel-major 384
  size_t o_k1d   = alloc((size_t)BN * CDIM * HWSZ * 2);   // channel-major 192
  size_t o_k2d   = alloc((size_t)BN * CDIM * HWSZ * 2);
  size_t o_v1T   = alloc(SZ_VT);
  size_t o_v2T   = alloc(SZ_VT);
  size_t o_creg  = alloc(3 * SZ_C2CM);                    // qc/k1c/k2c ; hh alias
  size_t o_oT    = alloc((size_t)BN * HWSZ * 384 * 2);    // N-major, ld=384
  size_t o_x2    = alloc((size_t)BN * CDIM * HWSZ * 4);
  size_t o_gT    = alloc((size_t)BN * HWSZ * HIDKP * 2);  // N-major, ld=512

  _Float16* qw16  = (_Float16*)(ws + o_qw16);
  _Float16* k1w16 = (_Float16*)(ws + o_k1w16);
  _Float16* k2w16 = (_Float16*)(ws + o_k2w16);
  _Float16* pjw16 = (_Float16*)(ws + o_pjw16);
  _Float16* pnw16 = (_Float16*)(ws + o_pnw16);
  _Float16* pow16 = (_Float16*)(ws + o_pow16);
  float*    invq  = (float*)(ws + o_invq);
  float*    invk  = (float*)(ws + o_invk);
  float*    Sbuf  = (float*)(ws + o_S);
  _Float16* A16   = (_Float16*)(ws + o_A16);
  _Float16* xnT   = (_Float16*)(ws + o_xnT);
  _Float16* k1nT  = (_Float16*)(ws + o_k1nT);
  _Float16* k2nT  = (_Float16*)(ws + o_k2nT);
  _Float16* qd16  = (_Float16*)(ws + o_qd);
  _Float16* k1d16 = (_Float16*)(ws + o_k1d);
  _Float16* k2d16 = (_Float16*)(ws + o_k2d);
  _Float16* v1T   = (_Float16*)(ws + o_v1T);
  _Float16* v2T   = (_Float16*)(ws + o_v2T);
  _Float16* qc16  = (_Float16*)(ws + o_creg);
  _Float16* k1c16 = (_Float16*)(ws + o_creg + SZ_C2CM);
  _Float16* k2c16 = (_Float16*)(ws + o_creg + 2 * SZ_C2CM);
  _Float16* hh16  = (_Float16*)(ws + o_creg);   // alias: c-buffers dead by pin GEMM
  _Float16* oT    = (_Float16*)(ws + o_oT);
  float*    x2    = (float*)(ws + o_x2);
  _Float16* gT    = (_Float16*)(ws + o_gT);
  _Float16* y16T  = xnT;                        // alias: xn dead after q GEMM

  // ---- 1) stage weights to f16 (zero padded) ----
  wconv_k<<<288, 256, 0, stream>>>(qw,   384, 192, qw16,  384, 192);
  wconv_k<<<288, 256, 0, stream>>>(kv1w, 384, 192, k1w16, 384, 192);
  wconv_k<<<288, 256, 0, stream>>>(kv2w, 384, 192, k2w16, 384, 192);
  wconv_k<<<288, 256, 0, stream>>>(projw, 192, 384, pjw16, 192, 384);
  wconv_k<<<768, 256, 0, stream>>>(pinw, HID2, 192, pnw16, HID2P, 192);
  wconv_k<<<384, 256, 0, stream>>>(poutw, 192, HIDN, pow16, 192, HIDKP);

  // zero V pad regions so apply_k's k in [48,64) reads exact 0.0 (never NaN)
  hipMemsetAsync(v1T, 0, SZ_VT, stream);
  hipMemsetAsync(v2T, 0, SZ_VT, stream);

  // ---- 2) channel LayerNorms -> transposed f16 ----
  ln_k<<<2048, 256, 0, stream>>>(x,      n1w,  n1b,  xnT,  CDIM);
  ln_k<<<2048, 256, 0, stream>>>(kv1_in, nk1w, nk1b, k1nT, CDIM);
  ln_k<<<2048, 256, 0, stream>>>(kv2_in, nk2w, nk2b, k2nT, CDIM);

  // ---- 3) 1x1 convs (WMMA GEMMs): q / kv1 / kv2 ----
  gemm_k<192, 6, false><<<dim3(128, 6, BN), 256, 0, stream>>>(
      qw16, xnT, qc16, nullptr, nullptr, 384);
  gemm_k<192, 6, false><<<dim3(128, 6, BN), 256, 0, stream>>>(
      k1w16, k1nT, k1c16, nullptr, nullptr, 384);
  gemm_k<192, 6, false><<<dim3(128, 6, BN), 256, 0, stream>>>(
      k2w16, k2nT, k2c16, nullptr, nullptr, 384);

  // ---- 4) grouped / depthwise 3x3 convs ----
  dwconv_q_k<<<98304, 256, 0, stream>>>(qc16, qdw, qd16);
  dwconv_kv_k<<<98304, 256, 0, stream>>>(k1c16, k1dw, k1d16, v1T);
  dwconv_kv_k<<<98304, 256, 0, stream>>>(k2c16, k2dw, k2d16, v2T);

  // ---- 5) channel attention ----
  rownorm_k<<<3072, 256, 0, stream>>>(qd16, k1d16, k2d16, invq, invk);
  gram_k<<<dim3(3, 3, 32), 256, 0, stream>>>(qd16, k1d16, k2d16, invq, invk,
                                             temp1, temp2, Sbuf);
  softmax_k<<<6, 256, 0, stream>>>(Sbuf, A16);
  apply_k<<<dim3(128, 3, 32), 256, 0, stream>>>(A16, v1T, v2T, oT);

  // ---- 6) proj + residual: x2 = x + proj(o) ----
  gemm_k<384, 12, true><<<dim3(128, 3, BN), 256, 0, stream>>>(
      pjw16, oT, nullptr, x2, x, CDIM);

  // ---- 7) FFN: LN2, pin, dw+gelu-gate, pout + residual -> d_out ----
  ln_k<<<2048, 256, 0, stream>>>(x2, n2w, n2b, y16T, CDIM);
  gemm_k<192, 6, false><<<dim3(128, 16, BN), 256, 0, stream>>>(
      pnw16, y16T, hh16, nullptr, nullptr, HID2P);
  ffngate_k<<<131072, 256, 0, stream>>>(hh16, dww, gT);
  gemm_k<512, 16, true><<<dim3(128, 3, BN), 256, 0, stream>>>(
      pow16, gT, nullptr, outp, x2, CDIM);
}
